// EncoderDecoder_no_NA_54408645706013
// MI455X (gfx1250) — matrix operand order repaired and checked by reading, not compiled, not run on hardware
//
#include <hip/hip_runtime.h>

// ---------------------------------------------------------------------------
// CDNA5 (gfx1250) implicit-GEMM conv/deconv using v_wmma_f32_16x16x32_bf16
// with Tensor-Data-Mover (tensor_load_to_lds) staging of the weight tile.
// bf16 multiply, f32 accumulate. Wave32; 8 waves (256 thr) per workgroup.
// K is tap-major / channel-minor: each 32-wide K step sits inside one (ky,kx)
// tap, so conv/deconv coordinate math hoists out of the channel loop.
// ---------------------------------------------------------------------------

typedef __attribute__((ext_vector_type(16))) __bf16       v16bf;
typedef __attribute__((ext_vector_type(8)))  float        v8f;
typedef __attribute__((ext_vector_type(8)))  unsigned int v8u;
typedef __attribute__((ext_vector_type(4)))  unsigned int v4u;
typedef __attribute__((ext_vector_type(8)))  int          v8i;
typedef __attribute__((ext_vector_type(4)))  int          v4i;

#define SA_STRIDE 40   // bf16 elems per sA row: 64B data + 16B pad = 80B
#define SB_STRIDE 72   // bf16 elems per sB row
#define SA_BYTES  (128 * SA_STRIDE * 2)       // 10240
#define SB_BYTES  (32 * SB_STRIDE * 2)        // 4608
#define LDS_BYTES (SA_BYTES + SB_BYTES)       // 14848

// --------------------------- weight repack ---------------------------------
// A[C_out][KK*CinPad] bf16 row-major, K index = (ky*Kk+kx)*CinPad + ic.
struct PackParams { int Cout, Cin, CinPad, Kk, deconv; };

__global__ void pack_weights(const float* __restrict__ w,
                             __bf16* __restrict__ A, PackParams p) {
  int rowLen = p.Kk * p.Kk * p.CinPad;
  int idx = blockIdx.x * blockDim.x + threadIdx.x;
  if (idx >= p.Cout * rowLen) return;
  int oc  = idx / rowLen;
  int kk  = idx - oc * rowLen;
  int tap = kk / p.CinPad;
  int ic  = kk - tap * p.CinPad;
  float v = 0.0f;
  if (ic < p.Cin) {
    int ky = tap / p.Kk, kx = tap - ky * p.Kk;
    v = p.deconv ? w[((ic * p.Cout + oc) * p.Kk + ky) * p.Kk + kx]
                 : w[((oc * p.Cin  + ic) * p.Kk + ky) * p.Kk + kx];
  }
  A[idx] = (__bf16)v;
}

// --------------------------- conv / deconv GEMM ----------------------------
struct ConvParams {
  int NB, Cin, CinPad, Hin, Win, Cout, Hout, Wout;
  int Kk, stride, pad, deconv;
  int Npos, HinWin, rowLen;   // rowLen = Kk*Kk*CinPad
};

__launch_bounds__(256)
__global__ void conv_wmma(const float* __restrict__ x,
                          const __bf16* __restrict__ A,
                          const float* __restrict__ bias,
                          float* __restrict__ out, ConvParams cp) {
  extern __shared__ __align__(16) char dynsmem[];
  __bf16* sA = (__bf16*)dynsmem;               // [128][SA_STRIDE]
  __bf16* sB = (__bf16*)(dynsmem + SA_BYTES);  // [32][SB_STRIDE]

  const int tid  = threadIdx.x;
  const int lane = tid & 31;
  const int wave = tid >> 5;
  const int wm   = (wave & 3) * 32;   // wave M offset inside 128-row tile
  const int wn   = (wave >> 2) * 32;  // wave N offset inside 64-col tile

  const int posBase = blockIdx.x * 64;
  const int ocBase  = blockIdx.y * 128;
  const int HW      = cp.Hout * cp.Wout;

  v8f acc[2][2] = {};

  // ---- per-thread B-gather geometry (one-time divisions) ----
  const int brow  = tid & 31;          // channel row within 32-step
  const int bcol0 = (tid >> 5) * 8;    // 8 columns per thread
  int xb0[8], poy[8], pox[8];
  bool pvalid[8];
  #pragma unroll
  for (int c = 0; c < 8; ++c) {
    int p = posBase + bcol0 + c;
    pvalid[c] = (p < cp.Npos);
    if (!pvalid[c]) p = 0;
    int n  = p / HW;
    int r  = p - n * HW;
    int oy = r / cp.Wout;
    poy[c] = oy;
    pox[c] = r - oy * cp.Wout;
    xb0[c] = n * cp.Cin * cp.HinWin;
  }

  // ---- TDM descriptor invariants for the A (weight) tile ----
  const unsigned long long aTileBase =
      (unsigned long long)(const void*)(A + (size_t)ocBase * cp.rowLen);
  const unsigned ldsA    = __builtin_amdgcn_groupstaticsize();  // dyn-LDS base
  const unsigned t0      = (unsigned)(cp.rowLen >> 2);  // tensor_dim0, 8B units
  const unsigned t1      = (unsigned)(cp.Cout > ocBase ? cp.Cout - ocBase : 0);
  const unsigned strideU = (unsigned)(cp.rowLen >> 2);  // row pitch, 8B units

#if !__has_builtin(__builtin_amdgcn_tensor_load_to_lds)
  // fallback manual A staging geometry
  const int  arow      = tid >> 1;
  const int  ahalf     = tid & 1;
  const bool arowValid = (ocBase + arow) < cp.Cout;
  const __bf16* aRowPtr = A + (size_t)(ocBase + arow) * cp.rowLen + ahalf * 16;
#endif

  const int chanSteps = cp.CinPad >> 5;   // 32-channel blocks per tap
  const int taps      = cp.Kk * cp.Kk;

  int ky = 0, kx = 0;
  for (int tap = 0; tap < taps; ++tap) {
    // ---- hoisted per-tap spatial geometry (no divisions) ----
    int xoff[8];
    bool ok[8];
    #pragma unroll
    for (int c = 0; c < 8; ++c) {
      int iy, ix; bool o;
      if (!cp.deconv) {
        iy = poy[c] * cp.stride - cp.pad + ky;
        ix = pox[c] * cp.stride - cp.pad + kx;
        o  = (iy >= 0) && (iy < cp.Hin) && (ix >= 0) && (ix < cp.Win);
      } else {
        int ty = poy[c] + cp.pad - ky;
        int tx = pox[c] + cp.pad - kx;
        o  = (ty >= 0) && (tx >= 0) && !(ty & (cp.stride - 1)) && !(tx & (cp.stride - 1));
        iy = ty / cp.stride; ix = tx / cp.stride;
        o  = o && (iy < cp.Hin) && (ix < cp.Win);
      }
      ok[c]   = o && pvalid[c];
      xoff[c] = o ? (iy * cp.Win + ix) : 0;
    }

    #pragma unroll 2
    for (int icb = 0; icb < chanSteps; ++icb) {
      const int k0 = tap * cp.CinPad + icb * 32;   // column into packed A
      __syncthreads();   // previous iteration's LDS reads complete

#if __has_builtin(__builtin_amdgcn_tensor_load_to_lds)
      // ---- TDM: DMA the 128x64B weight tile into LDS (wave 0 issues) ----
      // 2D tile, data_size=8B: tile_dim0=8 units (64B row), tile_dim1=128 rows,
      // row pitch = rowLen/4 units. Rows beyond Cout read zero (OOB).
      // pad: +4 DWORDs every 16 DWORDs -> LDS row stride 80B (= SA_STRIDE).
      if (wave == 0) {
        unsigned long long ga = aTileBase + (unsigned long long)(k0 * 2);
        v4u g0; v8i g1;
        v4i gz4 = {0, 0, 0, 0};
        v8i gz8 = {0, 0, 0, 0, 0, 0, 0, 0};
        g0[0] = 1u;                                   // count=1 (valid D#)
        g0[1] = ldsA;                                 // lds_addr (sA base)
        g0[2] = (unsigned)(ga & 0xFFFFFFFFull);       // global_addr[31:0]
        g0[3] = (unsigned)((ga >> 32) & 0x01FFFFFFull)
              | 0x80000000u;                          // addr[56:32] | type=2
        g1[0] = (3 << 16)                             // data_size = 8B
              | (1 << 20)                             // pad_enable
              | (3 << 22)                             // pad_interval = 16 DW
              | (3 << 25);                            // pad_amount   = 4 DW
        g1[1] = (int)((t0 & 0xFFFFu) << 16);          // tensor_dim0 lo16
        g1[2] = (int)((t0 >> 16) | ((t1 & 0xFFFFu) << 16)); // dim0 hi | dim1 lo
        g1[3] = (int)((t1 >> 16) | (8u << 16));       // dim1 hi | tile_dim0=8
        g1[4] = 128;                                  // tile_dim1 = 128 rows
        g1[5] = (int)strideU;                         // tensor_dim0_stride lo32
        g1[6] = 0;                                    // stride hi | dim1_stride
        g1[7] = 0;
        // 6-arg form (this toolchain): g0, g1, g2, g3, extra group, cpol
        __builtin_amdgcn_tensor_load_to_lds(g0, g1, gz4, gz4, gz8, 0);
      }
#else
      // ---- fallback: manual 128-bit staged copy of the A tile ----
      {
        uint4 lo = make_uint4(0u, 0u, 0u, 0u), hi = lo;
        if (arowValid) {
          const uint4* src = (const uint4*)(aRowPtr + k0);
          lo = src[0];
          hi = src[1];
        }
        *((uint4*)&sA[arow * SA_STRIDE + ahalf * 16 + 0]) = lo;
        *((uint4*)&sA[arow * SA_STRIDE + ahalf * 16 + 8]) = hi;
      }
#endif

      // ---- stage B tile (32x64 bf16): strided channel gather ----
      {
        const int ic     = icb * 32 + brow;
        const bool icOk  = ic < cp.Cin;
        const int icOff  = ic * cp.HinWin;
        #pragma unroll
        for (int c = 0; c < 8; ++c) {
          float v = 0.0f;
          if (icOk && ok[c])
            v = x[xb0[c] + icOff + xoff[c]];
          sB[brow * SB_STRIDE + bcol0 + c] = (__bf16)v;
        }
      }

#if __has_builtin(__builtin_amdgcn_tensor_load_to_lds)
      if (wave == 0) __builtin_amdgcn_s_wait_tensorcnt(0);  // TDM done
#endif
      __syncthreads();   // sA (TDM) + sB (ds) visible to all waves

      // ---- fragments per CDNA5 16-bit WMMA VGPR striping ----
      const int h     = lane >> 4;        // K-half select
      const int mrow0 = wm + (lane & 15);
      const int ncol0 = wn + (lane & 15);
      v16bf af[2], bfr[2];
      #pragma unroll
      for (int fi = 0; fi < 2; ++fi) {
        v8u tmp;
        const int row = mrow0 + fi * 16;
        #pragma unroll
        for (int v = 0; v < 8; ++v) {
          // A 16x32: lanes 0-15 K={0..7,16..23}, lanes 16-31 K={8..15,24..31}
          int kIdx = (v < 4) ? (h * 8 + v * 2) : (16 + h * 8 + (v - 4) * 2);
          tmp[v] = *((const unsigned*)&sA[row * SA_STRIDE + kIdx]);  // K,K+1
        }
        af[fi] = __builtin_bit_cast(v16bf, tmp);
      }
      #pragma unroll
      for (int fi = 0; fi < 2; ++fi) {
        v8u tmp;
        const int col = ncol0 + fi * 16;
        #pragma unroll
        for (int v = 0; v < 8; ++v) {
          // B 32x16: lanes 0-15 K={0..7,16..23}, lanes 16-31 K={8..15,24..31}
          int kIdx = (v < 4) ? (h * 8 + v * 2) : (16 + h * 8 + (v - 4) * 2);
          unsigned lo = (unsigned)__builtin_bit_cast(unsigned short,
                          sB[(kIdx + 0) * SB_STRIDE + col]);
          unsigned hi = (unsigned)__builtin_bit_cast(unsigned short,
                          sB[(kIdx + 1) * SB_STRIDE + col]);
          tmp[v] = lo | (hi << 16);
        }
        bfr[fi] = __builtin_bit_cast(v16bf, tmp);
      }

      #pragma unroll
      for (int i = 0; i < 2; ++i)
        #pragma unroll
        for (int j = 0; j < 2; ++j)
          acc[i][j] = __builtin_amdgcn_wmma_f32_16x16x32_bf16(
              false, af[i], false, bfr[j], (short)0, acc[i][j], false, false);
    }

    if (++kx == cp.Kk) { kx = 0; ++ky; }   // incremental tap coords
  }

  // ---- epilogue: C/D layout -> NCHW f32 store (+bias) ----
  const int ncol = lane & 15;
  const int mh   = lane >> 4;   // +8 rows for upper lane half
  #pragma unroll
  for (int i = 0; i < 2; ++i) {
    #pragma unroll
    for (int j = 0; j < 2; ++j) {
      int p = posBase + wn + j * 16 + ncol;
      if (p >= cp.Npos) continue;
      int n  = p / HW;
      int r  = p - n * HW;
      int oy = r / cp.Wout;
      int ox = r - oy * cp.Wout;
      #pragma unroll
      for (int v = 0; v < 8; ++v) {
        int oc = ocBase + wm + i * 16 + v + 8 * mh;
        if (oc < cp.Cout)
          out[(((size_t)n * cp.Cout + oc) * cp.Hout + oy) * cp.Wout + ox] =
              acc[i][j][v] + bias[oc];
      }
    }
  }
}

// --------------------------- host orchestration ----------------------------
struct LayerDef { int Cin, Hin, Win, Cout, Hout, Wout, Kk, stride, pad, deconv, wIdx, bIdx; };

extern "C" void kernel_launch(void* const* d_in, const int* in_sizes, int n_in,
                              void* d_out, int out_size, void* d_ws, size_t ws_size,
                              hipStream_t stream) {
  (void)in_sizes; (void)n_in; (void)out_size; (void)ws_size;
  const float* xin = (const float*)d_in[0];
  float* outp = (float*)d_out;
  char*  ws   = (char*)d_ws;
  const int NB = 4;

  const LayerDef L[12] = {
    {  3, 512, 512, 128, 256, 256, 5, 2, 2, 0,  1,  2},  // ga0
    {128, 256, 256, 192, 128, 128, 3, 2, 1, 0,  3,  4},  // ga2
    {192, 128, 128, 256,  64,  64, 3, 2, 1, 0,  5,  6},  // ga4
    {256,  64,  64, 320,  32,  32, 3, 2, 1, 0,  7,  8},  // ga6
    {320,  32,  32, 192,  16,  16, 3, 2, 1, 0,  9, 10},  // ha0
    {192,  16,  16, 192,   8,   8, 3, 2, 1, 0, 11, 12},  // ha2
    {192,   8,   8, 192,  16,  16, 3, 2, 1, 1, 13, 14},  // hs1
    {192,  16,  16, 320,  32,  32, 3, 2, 1, 1, 15, 16},  // hs3
    {320,  32,  32, 256,  64,  64, 3, 2, 1, 1, 17, 18},  // gs1
    {256,  64,  64, 192, 128, 128, 3, 2, 1, 1, 19, 20},  // gs3
    {192, 128, 128, 128, 256, 256, 3, 2, 1, 1, 21, 22},  // gs5
    {128, 256, 256,   3, 512, 512, 5, 2, 2, 1, 23, 24},  // gs7
  };

  size_t off = 0;
  auto carve = [&](size_t bytes) {
    size_t o = off;
    off = (off + bytes + 255) & ~(size_t)255;
    return o;
  };
  float* bufA = (float*)(ws + carve((size_t)NB * 128 * 256 * 256 * 4)); // 134MB max
  float* bufB = (float*)(ws + carve((size_t)NB * 192 * 128 * 128 * 4)); //  50MB max

  __bf16* packed[12];
  int cinPad[12];
  for (int i = 0; i < 12; ++i) {
    int cp32 = (L[i].Cin + 31) & ~31;
    cinPad[i] = cp32;
    size_t rowLen = (size_t)L[i].Kk * L[i].Kk * cp32;
    packed[i] = (__bf16*)(ws + carve((size_t)L[i].Cout * rowLen * 2));
  }

  for (int i = 0; i < 12; ++i) {
    PackParams pp{L[i].Cout, L[i].Cin, cinPad[i], L[i].Kk, L[i].deconv};
    int total = pp.Cout * pp.Kk * pp.Kk * pp.CinPad;
    pack_weights<<<(total + 255) / 256, 256, 0, stream>>>(
        (const float*)d_in[L[i].wIdx], packed[i], pp);
  }

  const float* src = xin;
  for (int i = 0; i < 12; ++i) {
    float* dst = (i == 11) ? outp : ((i % 2 == 0) ? bufA : bufB);
    ConvParams cp;
    cp.NB = NB;
    cp.Cin = L[i].Cin; cp.CinPad = cinPad[i];
    cp.Hin = L[i].Hin; cp.Win = L[i].Win;
    cp.Cout = L[i].Cout; cp.Hout = L[i].Hout; cp.Wout = L[i].Wout;
    cp.Kk = L[i].Kk; cp.stride = L[i].stride; cp.pad = L[i].pad;
    cp.deconv = L[i].deconv;
    cp.Npos   = NB * L[i].Hout * L[i].Wout;
    cp.HinWin = L[i].Hin * L[i].Win;
    cp.rowLen = L[i].Kk * L[i].Kk * cinPad[i];
    dim3 grid((cp.Npos + 63) / 64, (cp.Cout + 127) / 128);
    conv_wmma<<<grid, dim3(256), LDS_BYTES, stream>>>(
        src, packed[i], (const float*)d_in[L[i].bIdx], dst, cp);
    src = dst;
  }
}
